// Head_89644557402798
// MI455X (gfx1250) — matrix operand order repaired and checked by reading, not compiled, not run on hardware
//
#include <hip/hip_runtime.h>
#include <hip/hip_bf16.h>

typedef _Float16 f16;
typedef f16   v16h __attribute__((ext_vector_type(16)));
typedef f16   v8h  __attribute__((ext_vector_type(8)));
typedef f16   h2   __attribute__((ext_vector_type(2)));
typedef float v8f  __attribute__((ext_vector_type(8)));

#define BATCH 8
#define SEQ   2048
#define EMB   768
#define HD    64
#define BT    (BATCH*SEQ)

// ---- fragment loaders (CDNA5 16x16x32 f16 WMMA layouts, wave32) ----
// A (16x32, MxK): lane l -> row l%16; halves: K = kb + (l/16)*8 + i (i<8)
//                                      K = kb + 16 + (l/16)*8 + (i-8) (i>=8)
__device__ __forceinline__ v16h load_afrag(const f16* __restrict__ p, int ldm,
                                           int row, int kb, int half) {
  const v8h lo = *(const v8h*)(p + (size_t)row * ldm + kb + half * 8);
  const v8h hi = *(const v8h*)(p + (size_t)row * ldm + kb + 16 + half * 8);
  v16h a;
#pragma unroll
  for (int i = 0; i < 8; ++i) { a[i] = lo[i]; a[8 + i] = hi[i]; }
  return a;
}

// packed f16 max -> llvm.maxnum.v2f16 -> v_pk_max_num_f16
__device__ __forceinline__ h2 pkmax(h2 a, h2 b) {
  return __builtin_elementwise_max(a, b);
}

// ---- fp32 -> f16 conversion (grid-stride) ----
__global__ void cvt_f32_f16(const float* __restrict__ in, f16* __restrict__ out, int n) {
  int i = blockIdx.x * blockDim.x + threadIdx.x;
  int stride = gridDim.x * blockDim.x;
  for (; i < n; i += stride) out[i] = (f16)in[i];
}

// ---- projection: Y = X[BT,768] * W[768,64], f16 in, f16 out (f32 acc) ----
// TRANSPOSE=false: Y stored [BT, 64]   -> used for Q, V
// TRANSPOSE=true : Y stored [B, 64, T] -> used for K (contiguous K^T B-frags)
template <bool TRANSPOSE>
__global__ void __launch_bounds__(128) proj_wmma(const f16* __restrict__ x,
                                                 const f16* __restrict__ w,
                                                 f16* __restrict__ y) {
  const int wid  = threadIdx.x >> 5;
  const int lane = threadIdx.x & 31;
  const int half = lane >> 4;
  const int lr   = lane & 15;
  const int tile = blockIdx.x * 4 + wid;   // [0, BT/16)
  const int r0   = tile * 16;

  v8f acc[4] = {};
  for (int kb = 0; kb < EMB; kb += 32) {
    v16h a = load_afrag(x, EMB, r0 + lr, kb, half);
#pragma unroll
    for (int nt = 0; nt < 4; ++nt) {
      // B (32x16): lane l -> col N = nt*16+l%16; K = kb + (i&7) + 8*(l/16) + 16*(i/8)
      v16h b;
      for (int i = 0; i < 16; ++i) b[i] = w[(size_t)(kb + (i & 7) + 8 * half + 16 * (i >> 3)) * HD + nt * 16 + lr];
      acc[nt] = __builtin_amdgcn_wmma_f32_16x16x32_f16(false, a, false, b,
                                                       (short)0, acc[nt], false, false);
    }
  }
#pragma unroll
  for (int nt = 0; nt < 4; ++nt)
#pragma unroll
    for (int j = 0; j < 8; ++j) {
      const int r = r0 + j + 8 * half;          // global token index
      const int h = nt * 16 + lr;               // head-dim index
      if (TRANSPOSE) {
        const int b = r >> 11;                  // r / SEQ
        const int s = r & (SEQ - 1);            // tiles never straddle batches
        y[(size_t)b * HD * SEQ + (size_t)h * SEQ + s] = (f16)acc[nt][j];
      } else {
        y[(size_t)r * HD + h] = (f16)acc[nt][j];
      }
    }
}

// ---- one 32-wide s-block of online-softmax attention ----
// Q fragments pre-scaled by 0.125*log2(e): WMMA scores land in exp2-domain.
// Row-max reduction fully in packed f16 (v_cvt_pk_f16_f32 + v_pk_max_num_f16):
// rows (2t,2t+1) share one dword -> 16 ds_bpermute per block.
// Row-sum is deferred (per-lane partials).
template <bool MASKED>
__device__ __forceinline__ void attn_block(
    const int s0, const int q0, const int lane, const int half, const int lr,
    const v16h qa0, const v16h qa1,
    const f16* __restrict__ ktb, const f16* __restrict__ vp,
    f16* __restrict__ lp, float* __restrict__ mrow, float* __restrict__ lrow,
    v8f* __restrict__ o) {
  // ---- S = Q * K^T (16 x 32): contiguous B-frag rows of kt ----
  v8f sa = {}, sb = {};
#pragma unroll
  for (int hh = 0; hh < 2; ++hh) {
    const int h0 = hh * 32;
    v16h bka, bkb;
    for (int i = 0; i < 16; ++i) {
      const int kk = h0 + (i & 7) + 8 * half + 16 * (i >> 3);
      bka[i] = ktb[(size_t)kk * SEQ + s0 + lr];
      bkb[i] = ktb[(size_t)kk * SEQ + s0 + 16 + lr];
    }
    const v16h af  = hh ? qa1 : qa0;
    sa = __builtin_amdgcn_wmma_f32_16x16x32_f16(false, af, false, bka, (short)0, sa, false, false);
    sb = __builtin_amdgcn_wmma_f32_16x16x32_f16(false, af, false, bkb, (short)0, sb, false, false);
  }

  float ea[8], eb[8];
#pragma unroll
  for (int j = 0; j < 8; ++j) {
    ea[j] = sa[j];
    eb[j] = sb[j];
    if (MASKED) {
      const int qrow = q0 + j + 8 * half;
      ea[j] = (s0 + lr      > qrow) ? -INFINITY : ea[j];   // v_cndmask, EXEC untouched
      eb[j] = (s0 + 16 + lr > qrow) ? -INFINITY : eb[j];
    }
  }

  // ---- packed row-max reduction over the 16-lane half ----
  float mt[8];
#pragma unroll
  for (int t = 0; t < 4; ++t) {
    h2 a2, b2;
    a2[0] = (f16)ea[2 * t]; a2[1] = (f16)ea[2 * t + 1];   // v_cvt_pk_f16_f32
    b2[0] = (f16)eb[2 * t]; b2[1] = (f16)eb[2 * t + 1];
    h2 m2 = pkmax(a2, b2);
#pragma unroll
    for (int off = 8; off >= 1; off >>= 1) {
      int u = __shfl_xor(__builtin_bit_cast(int, m2), off, 32);
      m2 = pkmax(m2, __builtin_bit_cast(h2, u));
    }
    mt[2 * t]     = (float)m2[0];
    mt[2 * t + 1] = (float)m2[1];
  }

  float pa[8], pb[8], alpha[8];
#pragma unroll
  for (int j = 0; j < 8; ++j) {
    const float mnew = fmaxf(mrow[j], mt[j]);
    alpha[j] = exp2f(mrow[j] - mnew);
    const float p0 = exp2f(ea[j] - mnew);
    const float p1 = exp2f(eb[j] - mnew);
    lrow[j] = lrow[j] * alpha[j] + (p0 + p1);   // per-lane partial sum
    mrow[j] = mnew;
    pa[j] = p0; pb[j] = p1;
  }

  // ---- re-layout P: C-layout -> A-layout via per-wave LDS tile ----
#pragma unroll
  for (int j = 0; j < 8; ++j) {
    const int M = j + 8 * half;
    lp[M * 32 + lr]      = (f16)pa[j];
    lp[M * 32 + 16 + lr] = (f16)pb[j];
  }
  // same-wave LDS RAW: compiler inserts s_wait_dscnt
  const v16h pf = load_afrag(lp, 32, lr, 0, half);

  // ---- O = alpha*O + P * V  (16 x 64) ----
#pragma unroll
  for (int nt = 0; nt < 4; ++nt) {
#pragma unroll
    for (int j = 0; j < 8; ++j) o[nt][j] *= alpha[j];
    v16h bv;
    for (int i = 0; i < 16; ++i) bv[i] = vp[(size_t)(s0 + (i & 7) + 8 * half + 16 * (i >> 3)) * HD + nt * 16 + lr];
    o[nt] = __builtin_amdgcn_wmma_f32_16x16x32_f16(false, pf, false, bv, (short)0, o[nt], false, false);
  }
}

// ---- fused causal flash attention: one wave = 16 query rows x full H=64 ----
// kt is K transposed: [B, HD, SEQ]
__global__ void __launch_bounds__(128) flash_attn(const f16* __restrict__ q,
                                                  const f16* __restrict__ kt,
                                                  const f16* __restrict__ v,
                                                  float* __restrict__ out) {
  __shared__ __align__(16) f16 sP[4][16 * 32];   // per-wave P staging tile

  const int wid  = threadIdx.x >> 5;
  const int lane = threadIdx.x & 31;
  const int half = lane >> 4;
  const int lr   = lane & 15;

  const int g  = blockIdx.x * 4 + wid;    // [0, BATCH * SEQ/16)
  const int b  = g >> 7;                  // SEQ/16 == 128 q-tiles per batch
  const int q0 = (g & 127) * 16;

  const f16* qb  = q  + (size_t)b * SEQ * HD;
  const f16* ktb = kt + (size_t)b * HD * SEQ;
  const f16* vp  = v  + (size_t)b * SEQ * HD;
  f16* lp = sP[wid];

  // Q fragments, pre-scaled by (1/sqrt(H)) * log2(e) so scores are exp2-domain
  v16h qa0 = load_afrag(qb, HD, q0 + lr, 0, half);
  v16h qa1 = load_afrag(qb, HD, q0 + lr, 32, half);
  const f16 qsc = (f16)(0.125f * 1.44269504088896340736f);
#pragma unroll
  for (int i = 0; i < 16; ++i) { qa0[i] *= qsc; qa1[i] *= qsc; }

  float mrow[8], lrow[8];
  v8f o[4] = {};
#pragma unroll
  for (int j = 0; j < 8; ++j) { mrow[j] = -INFINITY; lrow[j] = 0.f; }

  // Main loop: blocks entirely below the diagonal (no masking).
  // Exactly one masked (diagonal) block always remains afterwards.
  int s0 = 0;
  for (; s0 + 31 <= q0; s0 += 32) {
    // prefetch next s-block of K^T and V (gfx1250 global_prefetch_b8);
    // the next block (possibly the diagonal one) always exists and is in-bounds
    __builtin_prefetch(ktb + (size_t)lane * SEQ + s0 + 32, 0, 1);
    __builtin_prefetch(ktb + (size_t)(32 + lane) * SEQ + s0 + 32, 0, 1);
    __builtin_prefetch(vp + (size_t)(s0 + 32 + lane) * HD, 0, 1);
    attn_block<false>(s0, q0, lane, half, lr, qa0, qa1, ktb, vp, lp, mrow, lrow, o);
  }
  // Diagonal block (the only one needing the causal mask)
  attn_block<true>(s0, q0, lane, half, lr, qa0, qa1, ktb, vp, lp, mrow, lrow, o);

  // ---- epilogue: deferred row-sum reduction (f32), normalize, store fp32 ----
  float inv[8];
#pragma unroll
  for (int j = 0; j < 8; ++j) {
    float rs = lrow[j];
#pragma unroll
    for (int off = 8; off >= 1; off >>= 1)
      rs += __shfl_xor(rs, off, 32);
    inv[j] = 1.0f / rs;
  }
  float* ob = out + (size_t)b * SEQ * HD;
#pragma unroll
  for (int nt = 0; nt < 4; ++nt)
#pragma unroll
    for (int j = 0; j < 8; ++j)
      ob[(size_t)(q0 + j + 8 * half) * HD + nt * 16 + lr] = o[nt][j] * inv[j];
}

extern "C" void kernel_launch(void* const* d_in, const int* in_sizes, int n_in,
                              void* d_out, int out_size, void* d_ws, size_t ws_size,
                              hipStream_t stream) {
  (void)in_sizes; (void)n_in; (void)out_size; (void)ws_size;
  const float* x  = (const float*)d_in[0];
  const float* Wk = (const float*)d_in[1];
  const float* Wq = (const float*)d_in[2];
  const float* Wv = (const float*)d_in[3];
  float* out = (float*)d_out;

  char* ws = (char*)d_ws;
  size_t off = 0;
  auto alloc = [&](size_t bytes) -> void* {
    void* p = ws + off;
    off = (off + bytes + 255) & ~(size_t)255;
    return p;
  };
  f16* xf  = (f16*)alloc((size_t)BT * EMB * sizeof(f16));
  f16* wkf = (f16*)alloc((size_t)EMB * HD * sizeof(f16));
  f16* wqf = (f16*)alloc((size_t)EMB * HD * sizeof(f16));
  f16* wvf = (f16*)alloc((size_t)EMB * HD * sizeof(f16));
  f16* ktf = (f16*)alloc((size_t)BT * HD * sizeof(f16));   // [B, HD, SEQ]
  f16* qf  = (f16*)alloc((size_t)BT * HD * sizeof(f16));   // [BT, HD]
  f16* vf  = (f16*)alloc((size_t)BT * HD * sizeof(f16));   // [BT, HD]

  // 1) cast inputs to f16
  cvt_f32_f16<<<2048, 256, 0, stream>>>(x,  xf,  BT * EMB);
  cvt_f32_f16<<<192,  256, 0, stream>>>(Wk, wkf, EMB * HD);
  cvt_f32_f16<<<192,  256, 0, stream>>>(Wq, wqf, EMB * HD);
  cvt_f32_f16<<<192,  256, 0, stream>>>(Wv, wvf, EMB * HD);

  // 2) K/Q/V projections via WMMA (1024 row-tiles, 4 waves/block)
  //    K is written pre-transposed so attention B-frags are contiguous.
  proj_wmma<true ><<<256, 128, 0, stream>>>(xf, wkf, ktf);
  proj_wmma<false><<<256, 128, 0, stream>>>(xf, wqf, qf);
  proj_wmma<false><<<256, 128, 0, stream>>>(xf, wvf, vf);

  // 3) fused causal flash attention (1024 q-tiles, 4 waves/block)
  flash_attn<<<256, 128, 0, stream>>>(qf, ktf, vf, out);
}
